// Qwen3_5GatedDeltaNet_11802570129918
// MI455X (gfx1250) — matrix operand order repaired and checked by reading, not compiled, not run on hardware
//
#include <hip/hip_runtime.h>

#define S_LEN 2048
#define HID 2048
#define NK 16
#define NV 32
#define DK 128
#define DV 128
#define KSZ 4
#define CHUNK 64
#define NCHUNK (S_LEN / CHUNK)
#define KEY_DIM (NK * DK)
#define VAL_DIM (NV * DV)
#define CONV_DIM (2 * KEY_DIM + VAL_DIM)

typedef __attribute__((ext_vector_type(16))) __bf16 v16bf;
typedef __attribute__((ext_vector_type(16))) unsigned short v16u;
typedef __attribute__((ext_vector_type(8)))  float v8f;

__device__ __forceinline__ unsigned short f2bf(float f) {
  union { float f; unsigned u; } c; c.f = f;
  unsigned r = c.u + 0x7fffu + ((c.u >> 16) & 1u);
  return (unsigned short)(r >> 16);
}
__device__ __forceinline__ float bf2f(unsigned short h) {
  union { unsigned u; float f; } c; c.u = ((unsigned)h) << 16;
  return c.f;
}

__device__ __forceinline__ v8f wmma_bf16(v16u a, v16u b, v8f c) {
#if defined(__HIP_DEVICE_COMPILE__)
  union { v16u u; v16bf b; } ca, cb;
  ca.u = a; cb.u = b;
  return __builtin_amdgcn_wmma_f32_16x16x32_bf16(false, ca.b, false, cb.b,
                                                 (short)0, c, false, false);
#else
  (void)a; (void)b; return c;
#endif
}

// gfx1250 async global->LDS copy (16B per lane), tracked by ASYNCcnt.
__device__ __forceinline__ void async_copy_b128(void* lds_dst, const void* gsrc) {
#if defined(__HIP_DEVICE_COMPILE__)
  unsigned lds_off = (unsigned)(unsigned long long)lds_dst;  // low 32b = LDS addr
  asm volatile("global_load_async_to_lds_b128 %0, %1, off"
               :: "v"(lds_off), "v"(gsrc) : "memory");
#else
  (void)lds_dst; (void)gsrc;
#endif
}
__device__ __forceinline__ void async_wait0() {
#if defined(__HIP_DEVICE_COMPILE__)
  asm volatile("s_wait_asynccnt 0x0" ::: "memory");
#endif
}

// ---- WMMA fragment loaders (CDNA5 ISA 7.12.2 layouts) -----------------------
__device__ __forceinline__ v16u load_fragA(const unsigned short* smem, int ld,
                                           int row0, int k0, int lane) {
  int r = row0 + (lane & 15);
  int kb = k0 + 8 * (lane >> 4);
  const unsigned short* p = smem + r * ld;
  v16u o;
#pragma unroll
  for (int e = 0; e < 8; ++e) o[e] = p[kb + e];
#pragma unroll
  for (int e = 0; e < 8; ++e) o[8 + e] = p[kb + 16 + e];
  return o;
}
__device__ __forceinline__ v16u load_fragA_rs(const unsigned short* smem, int ld,
                                              int row0, int k0, int lane, float s) {
  int r = row0 + (lane & 15);
  int kb = k0 + 8 * (lane >> 4);
  const unsigned short* p = smem + r * ld;
  v16u o;
#pragma unroll
  for (int e = 0; e < 8; ++e) o[e] = f2bf(bf2f(p[kb + e]) * s);
#pragma unroll
  for (int e = 0; e < 8; ++e) o[8 + e] = f2bf(bf2f(p[kb + 16 + e]) * s);
  return o;
}
__device__ __forceinline__ v16u load_fragAt_ks(const unsigned short* smem, int ld,
                                               int row0, int k0, int lane,
                                               const float* ks) {
  int r = row0 + (lane & 15);
  int kb = k0 + 8 * (lane >> 4);
  v16u o;
#pragma unroll
  for (int e = 0; e < 8; ++e) { int k = kb + e;      o[e]     = f2bf(bf2f(smem[k * ld + r]) * ks[k]); }
#pragma unroll
  for (int e = 0; e < 8; ++e) { int k = kb + 16 + e; o[8 + e] = f2bf(bf2f(smem[k * ld + r]) * ks[k]); }
  return o;
}
__device__ __forceinline__ v16u load_fragB(const unsigned short* smem, int ld,
                                           int row0, int k0, int lane) {
  int r = row0 + (lane & 15);
  int kb = k0 + 8 * (lane >> 4);
  const unsigned short* p = smem + r * ld + kb;
  v16u o;
#pragma unroll
  for (int e = 0; e < 16; ++e) o[e] = p[e + ((e >> 3) << 3)];
  return o;
}
__device__ __forceinline__ v16u load_fragBt(const unsigned short* smem, int ld,
                                            int n0, int k0, int lane) {
  int n = n0 + (lane & 15);
  int kb = k0 + 8 * (lane >> 4);
  v16u o;
#pragma unroll
  for (int e = 0; e < 16; ++e) o[e] = smem[(kb + e + ((e >> 3) << 3)) * ld + n];
  return o;
}

// ---- f32 -> bf16 bulk convert ----------------------------------------------
__global__ void f32_to_bf16(const float* __restrict__ in,
                            unsigned short* __restrict__ out, long n) {
  long i = ((long)blockIdx.x * 256 + threadIdx.x) * 4;
  if (i + 3 < n) {
    float4 v = *(const float4*)(in + i);
    out[i + 0] = f2bf(v.x); out[i + 1] = f2bf(v.y);
    out[i + 2] = f2bf(v.z); out[i + 3] = f2bf(v.w);
  }
}

// ---- GEMM: C[M,N] = A[M,K] @ B[N,K]^T ; A,B bf16 in global, C f32 -----------
// 128x128 block tile, double-buffered async global->LDS staging.
__global__ __launch_bounds__(256) void gemm_bt_bf(const unsigned short* __restrict__ A,
                                                  const unsigned short* __restrict__ B,
                                                  float* __restrict__ C,
                                                  int M, int N, int K) {
  __shared__ unsigned short sA[2][128][32];
  __shared__ unsigned short sB[2][128][32];
  int tid = threadIdx.x, lane = tid & 31, w = tid >> 5;
  int m0 = blockIdx.y * 128, n0 = blockIdx.x * 128;
  int mta = (w >> 1) * 2, ntb = (w & 1) * 4;

  auto issue = [&](int buf, int k0) {
    for (int j = tid; j < 1024; j += 256) {
      int isB = j >> 9, jj = j & 511;
      int r = jj >> 2, seg = (jj & 3) * 8;
      const unsigned short* g = isB ? (B + (size_t)(n0 + r) * K + k0 + seg)
                                    : (A + (size_t)(m0 + r) * K + k0 + seg);
      unsigned short* l = isB ? &sB[buf][r][seg] : &sA[buf][r][seg];
      async_copy_b128(l, g);
    }
  };

  v8f acc[2][4] = {};
  int nk = K / 32;
  issue(0, 0);
  for (int t = 0; t < nk; ++t) {
    async_wait0();
    __syncthreads();
    if (t + 1 < nk) issue((t + 1) & 1, (t + 1) * 32);
    int buf = t & 1;
    v16u af0 = load_fragA(&sA[buf][0][0], 32, mta * 16, 0, lane);
    v16u af1 = load_fragA(&sA[buf][0][0], 32, (mta + 1) * 16, 0, lane);
#pragma unroll
    for (int i = 0; i < 4; ++i) {
      v16u bf = load_fragB(&sB[buf][0][0], 32, (ntb + i) * 16, 0, lane);
      acc[0][i] = wmma_bf16(af0, bf, acc[0][i]);
      acc[1][i] = wmma_bf16(af1, bf, acc[1][i]);
    }
  }
  int col = lane & 15, rb = 8 * (lane >> 4);
#pragma unroll
  for (int mi = 0; mi < 2; ++mi)
#pragma unroll
    for (int i = 0; i < 4; ++i) {
      int cn = n0 + (ntb + i) * 16 + col;
      int rm = m0 + (mta + mi) * 16 + rb;
#pragma unroll
      for (int r = 0; r < 8; ++r) C[(size_t)(rm + r) * N + cn] = acc[mi][i][r];
    }
}

// ---- a,b projections --------------------------------------------------------
__global__ void ab_kernel(const float* __restrict__ x, const float* __restrict__ Wa,
                          const float* __restrict__ Wb, float* __restrict__ a,
                          float* __restrict__ b) {
  int s = blockIdx.x, t = threadIdx.x, col = t & 31;
  const float* W = (t < 32) ? Wa : Wb;
  float sum = 0.f;
  for (int k = 0; k < HID; ++k) sum += x[(size_t)s * HID + k] * W[(size_t)col * HID + k];
  if (t < 32) a[s * NV + col] = sum; else b[s * NV + col] = sum;
}

// ---- causal depthwise conv1d + SiLU -----------------------------------------
__global__ void conv_silu(const float* __restrict__ pre, const float* __restrict__ w,
                          float* __restrict__ post) {
  size_t i = (size_t)blockIdx.x * 256 + threadIdx.x;
  int s = (int)(i / CONV_DIM), c = (int)(i % CONV_DIM);
  float acc = 0.f;
#pragma unroll
  for (int t = 0; t < KSZ; ++t) {
    int sp = s - (KSZ - 1) + t;
    if (sp >= 0) acc += w[c * KSZ + t] * pre[(size_t)sp * CONV_DIM + c];
  }
  post[i] = acc / (1.f + expf(-acc));
}

// ---- q/k L2 norm (q also scaled by DK^-0.5) ---------------------------------
__global__ void qknorm(const float* __restrict__ mixed, float* __restrict__ qn,
                       float* __restrict__ kn) {
  int hk = blockIdx.x, s = blockIdx.y, d = threadIdx.x;
  float qv = mixed[(size_t)s * CONV_DIM + hk * DK + d];
  float kv = mixed[(size_t)s * CONV_DIM + KEY_DIM + hk * DK + d];
  __shared__ float rq[128], rk[128];
  rq[d] = qv * qv; rk[d] = kv * kv;
  __syncthreads();
  for (int o = 64; o; o >>= 1) {
    if (d < o) { rq[d] += rq[d + o]; rk[d] += rk[d + o]; }
    __syncthreads();
  }
  float qs = rsqrtf(rq[0] + 1e-6f) * 0.08838834764831845f;
  float ks = rsqrtf(rk[0] + 1e-6f);
  qn[((size_t)hk * S_LEN + s) * DK + d] = qv * qs;
  kn[((size_t)hk * S_LEN + s) * DK + d] = kv * ks;
}

// ---- gates ------------------------------------------------------------------
__global__ void gates_kernel(const float* __restrict__ a, const float* __restrict__ b,
                             const float* __restrict__ dt_bias, const float* __restrict__ A_log,
                             float* __restrict__ beta, float* __restrict__ graw) {
  int i = blockIdx.x * 256 + threadIdx.x;
  int s = i / NV, hv = i % NV;
  float bv = b[i];
  beta[i] = 1.f / (1.f + expf(-bv));
  float av = a[i] + dt_bias[hv];
  float sp = (av > 20.f) ? av : log1pf(expf(av));
  graw[(size_t)hv * S_LEN + s] = -expf(A_log[hv]) * sp;
}

__global__ void cumsum_kernel(const float* __restrict__ graw, float* __restrict__ gcum) {
  int i = blockIdx.x * 256 + threadIdx.x;
  if (i >= NV * NCHUNK) return;
  int hv = i / NCHUNK, n = i % NCHUNK;
  float c = 0.f;
  for (int j = 0; j < CHUNK; ++j) {
    c += graw[(size_t)hv * S_LEN + n * CHUNK + j];
    gcum[(size_t)hv * S_LEN + n * CHUNK + j] = c;
  }
}

// ---- per (head, chunk): attn build + UT transform + u/kcd -------------------
__global__ __launch_bounds__(256) void chunk_kernel(
    const float* __restrict__ kn, const float* __restrict__ mixed,
    const float* __restrict__ beta, const float* __restrict__ gcum,
    float* __restrict__ u, float* __restrict__ kcd) {
  __shared__ unsigned short sk[64][128];
  __shared__ unsigned short skb[64][128];
  __shared__ unsigned short svb[64][128];
  __shared__ float sattn[64][64];
  __shared__ unsigned short sattnb[64][64];
  __shared__ float sg[64];

  int n = blockIdx.x, hv = blockIdx.y, hk = hv >> 1;
  int tid = threadIdx.x, lane = tid & 31, w = tid >> 5;
  int s0 = n * CHUNK;

  if (tid < 64) sg[tid] = gcum[(size_t)hv * S_LEN + s0 + tid];
  for (int i = tid; i < 64 * 128; i += 256) {
    int r = i >> 7, c = i & 127;
    float kv = kn[((size_t)hk * S_LEN + s0 + r) * DK + c];
    float bt = beta[(s0 + r) * NV + hv];
    sk[r][c] = f2bf(kv);
    skb[r][c] = f2bf(kv * bt);
    float vv = mixed[(size_t)(s0 + r) * CONV_DIM + 2 * KEY_DIM + hv * DV + c];
    svb[r][c] = f2bf(vv * bt);
  }
  __syncthreads();

  for (int t = w * 2; t < w * 2 + 2; ++t) {
    int mt = t >> 2, nt = t & 3;
    v8f acc = {};
    for (int k0 = 0; k0 < 128; k0 += 32)
      acc = wmma_bf16(load_fragA(&skb[0][0], 128, mt * 16, k0, lane),
                      load_fragB(&sk[0][0], 128, nt * 16, k0, lane), acc);
    int col = nt * 16 + (lane & 15), rb = mt * 16 + 8 * (lane >> 4);
#pragma unroll
    for (int r = 0; r < 8; ++r) {
      int row = rb + r;
      sattn[row][col] = (row > col) ? -acc[r] * expf(sg[row] - sg[col]) : 0.f;
    }
  }
  __syncthreads();

  for (int i = 1; i < 64; ++i) {
    float sacc = 0.f;
    if (tid < 64)
      for (int j = 0; j < i; ++j) sacc += sattn[i][j] * sattn[j][tid];
    __syncthreads();
    if (tid < 64) sattn[i][tid] += sacc;
    __syncthreads();
  }
  for (int i = tid; i < 64 * 64; i += 256) {
    int r = i >> 6, c = i & 63;
    sattnb[r][c] = f2bf(sattn[r][c] + ((r == c) ? 1.f : 0.f));
  }
  for (int i = tid; i < 64 * 128; i += 256) {
    int r = i >> 7, c = i & 127;
    sk[r][c] = f2bf(bf2f(skb[r][c]) * expf(sg[r]));
  }
  __syncthreads();

  for (int t = w; t < 64; t += 8) {
    int which = t >> 5, tt = t & 31, mt = tt >> 3, nt = tt & 7;
    const unsigned short* Bm = which ? &sk[0][0] : &svb[0][0];
    v8f acc = {};
    for (int k0 = 0; k0 < 64; k0 += 32)
      acc = wmma_bf16(load_fragA(&sattnb[0][0], 64, mt * 16, k0, lane),
                      load_fragBt(Bm, 128, nt * 16, k0, lane), acc);
    float* dst = which ? kcd : u;
    int col = nt * 16 + (lane & 15), rb = mt * 16 + 8 * (lane >> 4);
#pragma unroll
    for (int r = 0; r < 8; ++r)
      dst[((size_t)hv * S_LEN + s0 + rb + r) * 128 + col] = acc[r];
  }
}

// ---- per head: sequential recurrence over chunks ----------------------------
__global__ __launch_bounds__(256) void step_kernel(
    const float* __restrict__ qn, const float* __restrict__ kn,
    const float* __restrict__ u, const float* __restrict__ kcd,
    const float* __restrict__ gcum, float* __restrict__ core) {
  __shared__ unsigned short sstate[128][128];
  __shared__ unsigned short sq[64][128];
  __shared__ unsigned short sk2[64][128];
  __shared__ unsigned short skcd[64][128];
  __shared__ unsigned short svn[64][128];
  __shared__ unsigned short sscore[64][64];
  __shared__ float sg2[64];
  __shared__ float sge[64];

  int hv = blockIdx.x, hk = hv >> 1;
  int tid = threadIdx.x, lane = tid & 31, w = tid >> 5;
  v8f st[8] = {};

  for (int n = 0; n < NCHUNK; ++n) {
    int s0 = n * CHUNK;
    if (tid < 64) sg2[tid] = gcum[(size_t)hv * S_LEN + s0 + tid];
    __syncthreads();
    float gl = sg2[63];
    if (tid < 64) sge[tid] = expf(gl - sg2[tid]);
    for (int i = tid; i < 64 * 128; i += 256) {
      int r = i >> 7, c = i & 127;
      sq[r][c] = f2bf(qn[((size_t)hk * S_LEN + s0 + r) * DK + c]);
      sk2[r][c] = f2bf(kn[((size_t)hk * S_LEN + s0 + r) * DK + c]);
      skcd[r][c] = f2bf(-kcd[((size_t)hv * S_LEN + s0 + r) * 128 + c]);
    }
    {
      int colb = lane & 15, rb = 8 * (lane >> 4);
#pragma unroll
      for (int t = 0; t < 8; ++t)
#pragma unroll
        for (int r = 0; r < 8; ++r)
          sstate[w * 16 + rb + r][t * 16 + colb] = f2bf(st[t][r]);
    }
    __syncthreads();

    for (int t = w * 2; t < w * 2 + 2; ++t) {
      int mt = t >> 2, nt = t & 3;
      v8f acc = {};
      for (int k0 = 0; k0 < 128; k0 += 32)
        acc = wmma_bf16(load_fragA(&sq[0][0], 128, mt * 16, k0, lane),
                        load_fragB(&sk2[0][0], 128, nt * 16, k0, lane), acc);
      int col = nt * 16 + (lane & 15), rb = mt * 16 + 8 * (lane >> 4);
#pragma unroll
      for (int r = 0; r < 8; ++r) {
        int row = rb + r;
        sscore[row][col] = f2bf((row >= col) ? acc[r] * expf(sg2[row] - sg2[col]) : 0.f);
      }
    }
    for (int t = w * 4; t < w * 4 + 4; ++t) {
      int mt = t >> 3, nt = t & 7;
      int col = nt * 16 + (lane & 15), rb = mt * 16 + 8 * (lane >> 4);
      v8f acc;
#pragma unroll
      for (int r = 0; r < 8; ++r)
        acc[r] = u[((size_t)hv * S_LEN + s0 + rb + r) * 128 + col];
      for (int k0 = 0; k0 < 128; k0 += 32)
        acc = wmma_bf16(load_fragA(&skcd[0][0], 128, mt * 16, k0, lane),
                        load_fragBt(&sstate[0][0], 128, nt * 16, k0, lane), acc);
#pragma unroll
      for (int r = 0; r < 8; ++r) svn[rb + r][col] = f2bf(acc[r]);
    }
    __syncthreads();

    for (int t = w * 4; t < w * 4 + 4; ++t) {
      int mt = t >> 3, nt = t & 7;
      float rowscale = expf(sg2[mt * 16 + (lane & 15)]);
      v8f acc = {};
      for (int k0 = 0; k0 < 128; k0 += 32)
        acc = wmma_bf16(load_fragA_rs(&sq[0][0], 128, mt * 16, k0, lane, rowscale),
                        load_fragBt(&sstate[0][0], 128, nt * 16, k0, lane), acc);
      for (int k0 = 0; k0 < 64; k0 += 32)
        acc = wmma_bf16(load_fragA(&sscore[0][0], 64, mt * 16, k0, lane),
                        load_fragBt(&svn[0][0], 128, nt * 16, k0, lane), acc);
      int col = nt * 16 + (lane & 15), rb = mt * 16 + 8 * (lane >> 4);
#pragma unroll
      for (int r = 0; r < 8; ++r)
        core[(size_t)(s0 + rb + r) * VAL_DIM + hv * DV + col] = acc[r];
    }
    float egl = expf(gl);
#pragma unroll
    for (int t = 0; t < 8; ++t) {
      v8f acc = st[t];
#pragma unroll
      for (int r = 0; r < 8; ++r) acc[r] *= egl;
      for (int k0 = 0; k0 < 64; k0 += 32)
        acc = wmma_bf16(load_fragAt_ks(&sk2[0][0], 128, w * 16, k0, lane, sge),
                        load_fragBt(&svn[0][0], 128, t * 16, k0, lane), acc);
      st[t] = acc;
    }
    __syncthreads();
  }
}

// ---- RMSNorm * norm_w * SiLU(z) -> bf16 h -----------------------------------
__global__ void rmsnorm_gate(const float* __restrict__ core, const float* __restrict__ z,
                             const float* __restrict__ norm_w,
                             unsigned short* __restrict__ hbf) {
  int row = blockIdx.x, d = threadIdx.x;
  float x = core[(size_t)row * DV + d];
  __shared__ float red[128];
  red[d] = x * x;
  __syncthreads();
  for (int o = 64; o; o >>= 1) {
    if (d < o) red[d] += red[d + o];
    __syncthreads();
  }
  float inv = rsqrtf(red[0] / (float)DV + 1e-6f);
  float zz = z[(size_t)row * DV + d];
  float h = x * inv * norm_w[d];
  hbf[(size_t)row * DV + d] = f2bf(h * (zz / (1.f + expf(-zz))));
}

extern "C" void kernel_launch(void* const* d_in, const int* in_sizes, int n_in,
                              void* d_out, int out_size, void* d_ws, size_t ws_size,
                              hipStream_t stream) {
  (void)in_sizes; (void)n_in; (void)out_size; (void)ws_size;
  const float* x       = (const float*)d_in[0];
  const float* W_qkv   = (const float*)d_in[1];
  const float* W_z     = (const float*)d_in[2];
  const float* W_a     = (const float*)d_in[3];
  const float* W_b     = (const float*)d_in[4];
  const float* conv_w  = (const float*)d_in[5];
  const float* dt_bias = (const float*)d_in[6];
  const float* A_log   = (const float*)d_in[7];
  const float* norm_w  = (const float*)d_in[8];
  const float* W_out   = (const float*)d_in[9];
  float* out = (float*)d_out;

  char* ws = (char*)d_ws;
  size_t off = 0;
  auto allocf = [&](size_t nel) { float* p = (float*)(ws + off); off += nel * 4; return p; };
  auto allocb = [&](size_t nel) { unsigned short* p = (unsigned short*)(ws + off); off += nel * 2; return p; };
  float* mixed_pre = allocf((size_t)S_LEN * CONV_DIM);
  float* mixed     = allocf((size_t)S_LEN * CONV_DIM);
  float* zbuf      = allocf((size_t)S_LEN * VAL_DIM);
  float* abuf      = allocf((size_t)S_LEN * NV);
  float* bbuf      = allocf((size_t)S_LEN * NV);
  float* qnorm_b   = allocf((size_t)NK * S_LEN * DK);
  float* knorm_b   = allocf((size_t)NK * S_LEN * DK);
  float* betab     = allocf((size_t)S_LEN * NV);
  float* graw      = allocf((size_t)NV * S_LEN);
  float* gcum      = allocf((size_t)NV * S_LEN);
  float* ubuf      = allocf((size_t)NV * S_LEN * DV);
  float* kcdbuf    = allocf((size_t)NV * S_LEN * DK);
  float* corebuf   = allocf((size_t)S_LEN * VAL_DIM);
  unsigned short* xbf    = allocb((size_t)S_LEN * HID);
  unsigned short* Wqkvbf = allocb((size_t)CONV_DIM * HID);
  unsigned short* Wzbf   = allocb((size_t)VAL_DIM * HID);
  unsigned short* Woutbf = allocb((size_t)HID * VAL_DIM);
  unsigned short* hbf    = allocb((size_t)S_LEN * VAL_DIM);

  auto cvt = [&](const float* src, unsigned short* dst, long n) {
    f32_to_bf16<<<(int)(n / 4 / 256), 256, 0, stream>>>(src, dst, n);
  };
  // 0. one-time bf16 conversions (halves all GEMM-side traffic)
  cvt(x, xbf, (long)S_LEN * HID);
  cvt(W_qkv, Wqkvbf, (long)CONV_DIM * HID);
  cvt(W_z, Wzbf, (long)VAL_DIM * HID);
  cvt(W_out, Woutbf, (long)HID * VAL_DIM);

  // 1. projections (async-staged bf16 WMMA GEMMs)
  gemm_bt_bf<<<dim3(CONV_DIM / 128, S_LEN / 128), 256, 0, stream>>>(xbf, Wqkvbf, mixed_pre, S_LEN, CONV_DIM, HID);
  gemm_bt_bf<<<dim3(VAL_DIM / 128, S_LEN / 128), 256, 0, stream>>>(xbf, Wzbf, zbuf, S_LEN, VAL_DIM, HID);
  ab_kernel<<<S_LEN, 64, 0, stream>>>(x, W_a, W_b, abuf, bbuf);
  // 2. conv + silu
  conv_silu<<<(int)((size_t)S_LEN * CONV_DIM / 256), 256, 0, stream>>>(mixed_pre, conv_w, mixed);
  // 3. q/k norm, gates, cumsum
  qknorm<<<dim3(NK, S_LEN), 128, 0, stream>>>(mixed, qnorm_b, knorm_b);
  gates_kernel<<<(S_LEN * NV) / 256, 256, 0, stream>>>(abuf, bbuf, dt_bias, A_log, betab, graw);
  cumsum_kernel<<<(NV * NCHUNK + 255) / 256, 256, 0, stream>>>(graw, gcum);
  // 4. chunk-parallel pre-pass
  chunk_kernel<<<dim3(NCHUNK, NV), 256, 0, stream>>>(knorm_b, mixed, betab, gcum, ubuf, kcdbuf);
  // 5. sequential recurrence, one persistent block per head
  step_kernel<<<NV, 256, 0, stream>>>(qnorm_b, knorm_b, ubuf, kcdbuf, gcum, corebuf);
  // 6. norm + gate (emit bf16 h), then output projection
  rmsnorm_gate<<<S_LEN * NV, 128, 0, stream>>>(corebuf, zbuf, norm_w, hbf);
  gemm_bt_bf<<<dim3(HID / 128, S_LEN / 128), 256, 0, stream>>>(hbf, Woutbf, out, S_LEN, HID, VAL_DIM);
}
